// BrosAttention_64544768524551
// MI455X (gfx1250) — matrix operand order repaired and checked by reading, not compiled, not run on hardware
//
#include <hip/hip_runtime.h>
#include <hip/hip_bf16.h>

// ---------------------------------------------------------------------------
// BrosAttention fused pipeline for MI455X (gfx1250, wave32, WMMA).
// B=2 S=1024 H=768 NH=12 DH=64.
// Memory-bound on the 512MB bbox_pos_emb read (~22us @ 23.3TB/s); all matmuls
// use v_wmma_f32_16x16x32_bf16 (bf16 in, fp32 accumulate). Fragment loads are
// vectorized b128 loads; fp32->bf16 uses native hardware converts.
// ---------------------------------------------------------------------------

typedef __attribute__((ext_vector_type(16))) __bf16 v16bf;
typedef __attribute__((ext_vector_type(8)))  __bf16 v8bf;
typedef __attribute__((ext_vector_type(4)))  __bf16 v4bf;
typedef __attribute__((ext_vector_type(8)))  float  v8f;

__device__ __forceinline__ v8f wmma_bf16(v16bf a, v16bf b, v8f c) {
    // (neg_a, A, neg_b, B, c_mod, C, reuse_a, reuse_b)
    return __builtin_amdgcn_wmma_f32_16x16x32_bf16(false, a, false, b, (short)0, c, false, false);
}

__device__ __forceinline__ v16bf combine(v8bf lo, v8bf hi) {
    return __builtin_shufflevector(lo, hi, 0, 1, 2, 3, 4, 5, 6, 7, 8, 9, 10, 11, 12, 13, 14, 15);
}
__device__ __forceinline__ v8bf ld8bf(const __bf16* p) { return *(const v8bf*)p; }

// 8 contiguous fp32 -> 8 bf16 (native hardware converts, 16B-aligned loads).
__device__ __forceinline__ v8bf cvt8(const float* p) {
    float4 f0 = *(const float4*)(p);
    float4 f1 = *(const float4*)(p + 4);
    v8bf r;
    r[0] = (__bf16)f0.x; r[1] = (__bf16)f0.y; r[2] = (__bf16)f0.z; r[3] = (__bf16)f0.w;
    r[4] = (__bf16)f1.x; r[5] = (__bf16)f1.y; r[6] = (__bf16)f1.z; r[7] = (__bf16)f1.w;
    return r;
}

// A fragment (16x32 bf16): lane row = lane&15; lanes 0-15 hold K=0..7,16..23;
// lanes 16-31 hold K=8..15,24..31. Row-major source, stride lda elements.
__device__ __forceinline__ v16bf load_a_bf(const __bf16* base, int lda, int k0) {
    int lane = threadIdx.x & 31;
    const __bf16* p = base + (size_t)(lane & 15) * lda + k0 + ((lane >> 4) * 8);
    return combine(ld8bf(p), ld8bf(p + 16));
}
__device__ __forceinline__ v16bf load_a_f32(const float* base, int lda, int k0) {
    int lane = threadIdx.x & 31;
    const float* p = base + (size_t)(lane & 15) * lda + k0 + ((lane >> 4) * 8);
    return combine(cvt8(p), cvt8(p + 16));
}

// B fragment (32x16 bf16): lane col = lane&15; lanes 0-15 hold K=0..7,16..23;
// lanes 16-31 hold K=8..15,24..31. Element B[k][col] at base + col*colStride + k.
__device__ __forceinline__ v16bf load_b_bf(const __bf16* base, int colStride, int k0) {
    int lane = threadIdx.x & 31;
    const __bf16* p = base + (size_t)(lane & 15) * colStride + k0 + ((lane >> 4) * 8);
    return combine(ld8bf(p), ld8bf(p + 16));
}
__device__ __forceinline__ v16bf load_b_f32(const float* base, int colStride, int k0) {
    int lane = threadIdx.x & 31;
    const float* p = base + (size_t)(lane & 15) * colStride + k0 + ((lane >> 4) * 8);
    return combine(cvt8(p), cvt8(p + 16));
}

// ---------------------------------------------------------------------------
// K1: fused Q/K/V projection. GEMM [2048x768] x [768x768]^T, 8 waves/block,
// one 16x16 tile per wave. Q,K stored bf16 [B,NH,S,DH]; V stored transposed
// bf16 [B,NH,DH,S] so PV B-fragments are K-contiguous.
// ---------------------------------------------------------------------------
__global__ void qkv_proj_k(const float* __restrict__ X,
                           const float* __restrict__ Wq, const float* __restrict__ bq,
                           const float* __restrict__ Wk, const float* __restrict__ bk,
                           const float* __restrict__ Wv, const float* __restrict__ bv,
                           __bf16* __restrict__ q_bf,
                           __bf16* __restrict__ k_bf,
                           __bf16* __restrict__ vT_bf) {
    int t   = blockIdx.x * 8 + (threadIdx.x >> 5);
    int mat = t / (128 * 48);
    int rem = t % (128 * 48);
    int mt = rem / 48, nt = rem % 48;
    const float* W    = (mat == 0) ? Wq : (mat == 1) ? Wk : Wv;
    const float* bias = (mat == 0) ? bq : (mat == 1) ? bk : bv;
    const float* ab = X + (size_t)(mt * 16) * 768;
    const float* bb = W + (size_t)(nt * 16) * 768;   // B[k][n] = W[n*768+k]
    v8f acc = {};
#pragma unroll 4
    for (int k0 = 0; k0 < 768; k0 += 32)
        acc = wmma_bf16(load_a_f32(ab, 768, k0), load_b_f32(bb, 768, k0), acc);
    int lane = threadIdx.x & 31, col = lane & 15, hi = lane >> 4;
    int ng = nt * 16 + col;
    float bvv = bias[ng];
    int h = ng >> 6, d = ng & 63;
#pragma unroll
    for (int r = 0; r < 8; ++r) {
        int mg = mt * 16 + r + 8 * hi;
        int bb2 = mg >> 10, s = mg & 1023;
        __bf16 val = (__bf16)(acc[r] + bvv);
        if (mat == 2)      vT_bf[((size_t)(bb2 * 12 + h) * 64 + d) * 1024 + s] = val;
        else if (mat == 1) k_bf [((size_t)(bb2 * 12 + h) * 1024 + s) * 64 + d] = val;
        else               q_bf [((size_t)(bb2 * 12 + h) * 1024 + s) * 64 + d] = val;
    }
}

// ---------------------------------------------------------------------------
// K2a: raw QK^T for one 128-row i-chunk. Wave keeps its Q fragments in
// registers and sweeps 8 j-tiles (16 WMMAs). scores[b,n,il,j] fp32 strip.
// ---------------------------------------------------------------------------
__global__ void scores_qk_k(const __bf16* __restrict__ q_bf,
                            const __bf16* __restrict__ k_bf,
                            float* __restrict__ scores, int i0) {
    int t  = blockIdx.x * 8 + (threadIdx.x >> 5);
    int bn = t / 64, rem = t % 64;
    int it = rem / 8, jg = rem % 8;
    const __bf16* qb = q_bf + ((size_t)bn * 1024 + i0 + it * 16) * 64;
    v16bf a0 = load_a_bf(qb, 64, 0);
    v16bf a1 = load_a_bf(qb, 64, 32);
    int lane = threadIdx.x & 31, col = lane & 15, hi = lane >> 4;
#pragma unroll
    for (int j2 = 0; j2 < 8; ++j2) {
        int jt = jg * 8 + j2;
        const __bf16* kb = k_bf + ((size_t)bn * 1024 + jt * 16) * 64;  // B[d][j]=k[j][d]
        v8f acc = {};
        acc = wmma_bf16(a0, load_b_bf(kb, 64, 0),  acc);
        acc = wmma_bf16(a1, load_b_bf(kb, 64, 32), acc);
        float* out = scores + ((size_t)bn * 128 + it * 16) * 1024 + jt * 16 + col;
#pragma unroll
        for (int r = 0; r < 8; ++r) out[(size_t)(r + 8 * hi) * 1024] = acc[r];
    }
}

// ---------------------------------------------------------------------------
// K2b: bbox bias. Per (b,i): A = q[b,heads,i,:] (12 rows padded to 16,
// loaded once) x B = bpe[b,i,jtile,:]^T, sweeping 8 j-tiles per wave. This is
// the single pass over the 512MB bpe tensor (roofline limiter); next tile is
// prefetched (global_prefetch_b8). Accumulates into the score strip.
// ---------------------------------------------------------------------------
__global__ void bias_add_k(const __bf16* __restrict__ q_bf,
                           const float* __restrict__ bpe,
                           float* __restrict__ scores, int i0) {
    int t = blockIdx.x * 8 + (threadIdx.x >> 5);
    int b = t / (128 * 8), rem = t % (128 * 8);
    int il = rem / 8, jg = rem % 8;
    int i  = i0 + il;
    int lane = threadIdx.x & 31;
    int row = lane & 15, kbq = (lane >> 4) * 8;
    int n = (row < 12) ? row : 0;                       // pad heads 12..15 (rows discarded)
    const __bf16* qp = q_bf + ((size_t)(b * 12 + n) * 1024 + i) * 64 + kbq;
    v16bf a0 = combine(ld8bf(qp),      ld8bf(qp + 16));
    v16bf a1 = combine(ld8bf(qp + 32), ld8bf(qp + 48));
    int col = lane & 15, hi = lane >> 4;
#pragma unroll
    for (int j2 = 0; j2 < 8; ++j2) {
        int jt = jg * 8 + j2;
        // bpe layout (S,S,B,DH): column d-vectors contiguous.
        const float* bb = bpe + (((size_t)i * 1024 + jt * 16) * 2 + b) * 64;
        __builtin_prefetch(bb + 2048, 0, 1);            // next j-tile of the bpe stream
        v8f acc = {};
        acc = wmma_bf16(a0, load_b_f32(bb, 128, 0),  acc);
        acc = wmma_bf16(a1, load_b_f32(bb, 128, 32), acc);
#pragma unroll
        for (int r = 0; r < 8; ++r) {
            int n2 = r + 8 * hi;
            if (n2 < 12) {
                float* sp = scores + ((size_t)(b * 12 + n2) * 128 + il) * 1024 + jt * 16 + col;
                *sp += acc[r];                          // unique owner per element
            }
        }
    }
}

// ---------------------------------------------------------------------------
// K3: softmax over j (row length 1024), scale 1/sqrt(64)=0.125; writes bf16.
// ---------------------------------------------------------------------------
__global__ void softmax_k(const float* __restrict__ scores,
                          __bf16* __restrict__ probs) {
    __shared__ float red[256];
    int row = blockIdx.x, t = threadIdx.x;
    float4 x4 = *(const float4*)(scores + (size_t)row * 1024 + t * 4);
    float x0 = x4.x * 0.125f, x1 = x4.y * 0.125f, x2 = x4.z * 0.125f, x3 = x4.w * 0.125f;
    float m = fmaxf(fmaxf(x0, x1), fmaxf(x2, x3));
    red[t] = m; __syncthreads();
    for (int off = 128; off > 0; off >>= 1) { if (t < off) red[t] = fmaxf(red[t], red[t + off]); __syncthreads(); }
    m = red[0]; __syncthreads();
    float e0 = __expf(x0 - m), e1 = __expf(x1 - m), e2 = __expf(x2 - m), e3 = __expf(x3 - m);
    red[t] = e0 + e1 + e2 + e3; __syncthreads();
    for (int off = 128; off > 0; off >>= 1) { if (t < off) red[t] += red[t + off]; __syncthreads(); }
    float inv = 1.0f / red[0];
    v4bf o;
    o[0] = (__bf16)(e0 * inv); o[1] = (__bf16)(e1 * inv);
    o[2] = (__bf16)(e2 * inv); o[3] = (__bf16)(e3 * inv);
    *(v4bf*)(probs + (size_t)row * 1024 + t * 4) = o;
}

// ---------------------------------------------------------------------------
// K4: ctx = probs @ V per (b,n). M=128 chunk rows, N=64, K=1024. V^T layout
// makes B-fragment loads K-contiguous. ctx stored [B,S,H] fp32 for out-proj.
// ---------------------------------------------------------------------------
__global__ void pv_k(const __bf16* __restrict__ probs,
                     const __bf16* __restrict__ vT_bf,
                     float* __restrict__ ctx, int i0) {
    int t  = blockIdx.x * 8 + (threadIdx.x >> 5);
    int bn = t / 32, rem = t % 32;
    int it = rem / 4, dt = rem % 4;
    const __bf16* ab = probs + ((size_t)bn * 128 + it * 16) * 1024;
    const __bf16* bb = vT_bf + ((size_t)bn * 64 + dt * 16) * 1024;
    v8f acc = {};
#pragma unroll 4
    for (int k0 = 0; k0 < 1024; k0 += 32)
        acc = wmma_bf16(load_a_bf(ab, 1024, k0), load_b_bf(bb, 1024, k0), acc);
    int lane = threadIdx.x & 31, col = lane & 15, hi = lane >> 4;
    int b = bn / 12, n = bn % 12;
#pragma unroll
    for (int r = 0; r < 8; ++r) {
        int ig = i0 + it * 16 + r + 8 * hi;
        ctx[(size_t)(b * 1024 + ig) * 768 + n * 64 + dt * 16 + col] = acc[r];
    }
}

// ---------------------------------------------------------------------------
// K5: out projection + residual: y = ctx @ Wo^T + bo + hidden (fp32 strip).
// ---------------------------------------------------------------------------
__global__ void out_proj_k(const float* __restrict__ ctx, const float* __restrict__ Wo,
                           const float* __restrict__ bo, const float* __restrict__ X,
                           float* __restrict__ y) {
    int t  = blockIdx.x * 8 + (threadIdx.x >> 5);
    int mt = t / 48, nt = t % 48;
    const float* ab = ctx + (size_t)(mt * 16) * 768;
    const float* bb = Wo  + (size_t)(nt * 16) * 768;
    v8f acc = {};
#pragma unroll 4
    for (int k0 = 0; k0 < 768; k0 += 32)
        acc = wmma_bf16(load_a_f32(ab, 768, k0), load_b_f32(bb, 768, k0), acc);
    int lane = threadIdx.x & 31, col = lane & 15, hi = lane >> 4;
    int ng = nt * 16 + col;
    float bvv = bo[ng];
#pragma unroll
    for (int r = 0; r < 8; ++r) {
        int mg = mt * 16 + r + 8 * hi;
        size_t idx = (size_t)mg * 768 + ng;
        y[idx] = acc[r] + bvv + X[idx];
    }
}

// ---------------------------------------------------------------------------
// K6: residual layernorm over H=768 (3 elements/thread, 256 threads).
// ---------------------------------------------------------------------------
__global__ void ln_k(const float* __restrict__ y, const float* __restrict__ g,
                     const float* __restrict__ be, float* __restrict__ out) {
    __shared__ float red[256];
    int row = blockIdx.x, t = threadIdx.x;
    const float* yr = y + (size_t)row * 768 + t * 3;
    float a0 = yr[0], a1 = yr[1], a2 = yr[2];
    red[t] = a0 + a1 + a2; __syncthreads();
    for (int off = 128; off > 0; off >>= 1) { if (t < off) red[t] += red[t + off]; __syncthreads(); }
    float mu = red[0] * (1.0f / 768.0f); __syncthreads();
    float d0 = a0 - mu, d1 = a1 - mu, d2 = a2 - mu;
    red[t] = d0 * d0 + d1 * d1 + d2 * d2; __syncthreads();
    for (int off = 128; off > 0; off >>= 1) { if (t < off) red[t] += red[t + off]; __syncthreads(); }
    float inv = rsqrtf(red[0] * (1.0f / 768.0f) + 1e-12f);
    float* o = out + (size_t)row * 768 + t * 3;
    const float* gp = g + t * 3; const float* bp = be + t * 3;
    o[0] = d0 * inv * gp[0] + bp[0];
    o[1] = d1 * inv * gp[1] + bp[1];
    o[2] = d2 * inv * gp[2] + bp[2];
}

// ---------------------------------------------------------------------------
// Host: workspace carve (34.6 MB) + stream-ordered launches. Score strip of
// 128 query rows is recycled over 8 chunks; y-buffer reuses the score strip.
// ---------------------------------------------------------------------------
extern "C" void kernel_launch(void* const* d_in, const int* in_sizes, int n_in,
                              void* d_out, int out_size, void* d_ws, size_t ws_size,
                              hipStream_t stream) {
    const float* hidden = (const float*)d_in[0];
    const float* bpe    = (const float*)d_in[1];
    const float* Wq = (const float*)d_in[2];  const float* bq = (const float*)d_in[3];
    const float* Wk = (const float*)d_in[4];  const float* bk = (const float*)d_in[5];
    const float* Wv = (const float*)d_in[6];  const float* bv = (const float*)d_in[7];
    const float* Wo = (const float*)d_in[8];  const float* bo = (const float*)d_in[9];
    const float* lng = (const float*)d_in[10];
    const float* lnb = (const float*)d_in[11];
    float* out = (float*)d_out;

    const size_t QKV_ELEMS   = (size_t)24 * 1024 * 64;   // B*NH*S*DH
    const size_t STRIP_ELEMS = (size_t)24 * 128 * 1024;  // B*NH*128*S

    __bf16* q_bf  = (__bf16*)d_ws;
    __bf16* k_bf  = q_bf + QKV_ELEMS;
    __bf16* vT_bf = k_bf + QKV_ELEMS;
    __bf16* probs = vT_bf + QKV_ELEMS;                   // bf16 strip
    float* scores = (float*)(probs + STRIP_ELEMS);       // fp32 strip (12.6 MB)
    float* ctx    = scores + STRIP_ELEMS;                // [B,S,H] fp32
    float* ybuf   = scores;                              // reuse strip after chunks

    // K1: 3 mats x 128 x 48 tiles, 8 waves/block
    qkv_proj_k<<<2304, 256, 0, stream>>>(hidden, Wq, bq, Wk, bk, Wv, bv, q_bf, k_bf, vT_bf);

    for (int c = 0; c < 8; ++c) {
        int i0 = c * 128;
        scores_qk_k<<< 192, 256, 0, stream>>>(q_bf, k_bf, scores, i0);   // 24*8*8 waves
        bias_add_k <<< 256, 256, 0, stream>>>(q_bf, bpe, scores, i0);    // 2*128*8 waves
        softmax_k  <<<3072, 256, 0, stream>>>(scores, probs);            // 24*128 rows
        pv_k       <<<  96, 256, 0, stream>>>(probs, vT_bf, ctx, i0);    // 24*8*4 tiles
    }

    out_proj_k<<<768, 256, 0, stream>>>(ctx, Wo, bo, hidden, ybuf);      // 128*48 tiles
    ln_k      <<<2048, 256, 0, stream>>>(ybuf, lng, lnb, out);           // 2048 rows
    (void)in_sizes; (void)n_in; (void)out_size; (void)ws_size;
}
